// Quantizer_6554120093898
// MI455X (gfx1250) — matrix operand order repaired and checked by reading, not compiled, not run on hardware
//
#include <hip/hip_runtime.h>
#include <hip/hip_bf16.h>

// ---------------------------------------------------------------------------
// VQ quantizer for MI455X (gfx1250, wave32, WMMA).
// B=16, C=64, H=64, W=64 -> N = 65536 tokens; K = 1024 codes.
// dist argmin uses  min_k( e2[k] - 2 * x . e_k )  via bf16x2-split WMMA.
// ---------------------------------------------------------------------------

#define NB   16
#define NC   64
#define NH   64
#define NW   64
#define NTOK 65536      /* NB*NH*NW */
#define NK   1024
#define NELE 4194304    /* NTOK*NC  */

typedef __attribute__((ext_vector_type(16))) __bf16 v16bf;
typedef __attribute__((ext_vector_type(8)))  float  v8f;

union BV {               // one WMMA 16-bit operand (8 VGPRs per lane)
    uint4  q[2];
    v16bf  v;
    unsigned short s[16];
};

__device__ __forceinline__ void split_bf16(float f, unsigned short& hi, unsigned short& lo) {
    unsigned u = __float_as_uint(f);
    hi = (unsigned short)(u >> 16);                       // truncated high bf16
    float hif = __uint_as_float(u & 0xffff0000u);
    float lof = f - hif;                                  // exact residual
    lo = (unsigned short)(__float_as_uint(lof) >> 16);    // ~2^-16 rel. total
}

// ---------------------------------------------------------------------------
// Prep 1: split codebook into hi/lo bf16, pre-swizzled into WMMA-B lane order.
// Dest layout (half-index): flat = tile*1024 + seg*256 + lane*8 + w
//   seg(0..3): two K=32 chunks x two 16B pieces; per-lane 16B => conflict-free.
// B element (K=c_local, N): lane = (N%16) + 16*(c_local/16),
//   within lane: vgpr v holds K = 16*(lane/16) + 2v + {0,1} (chunk h adds 32h).
// ---------------------------------------------------------------------------
__global__ __launch_bounds__(256) void vq_prep_split(const float* __restrict__ emb,
                                                     unsigned short* __restrict__ bhi,
                                                     unsigned short* __restrict__ blo) {
    int flat = blockIdx.x * 256 + threadIdx.x;            // 0 .. 65535
    int tile = flat >> 10;
    int rem  = flat & 1023;
    int seg  = rem >> 8;                                  // 0..3
    int lane = (rem >> 3) & 31;
    int w    = rem & 7;
    int h    = seg >> 1;                                  // K-chunk (channels 32h..)
    int j    = ((seg & 1) << 3) | w;                      // 0..15 within chunk
    int g    = lane >> 4;
    int v    = j >> 1;
    int t    = j & 1;
    int c    = (h << 5) + (g << 4) + (v << 1) + t;        // channel 0..63
    int n    = (tile << 4) + (lane & 15);                 // code id
    float val = emb[n * NC + c];
    unsigned short hi, lo;
    split_bf16(val, hi, lo);
    bhi[flat] = hi;
    blo[flat] = lo;
}

// Prep 2: e2[k] = sum_c emb[k,c]^2
__global__ __launch_bounds__(256) void vq_prep_e2(const float* __restrict__ emb,
                                                  float* __restrict__ e2) {
    int k = blockIdx.x * 256 + threadIdx.x;
    if (k < NK) {
        float s = 0.f;
        #pragma unroll 8
        for (int c = 0; c < NC; ++c) { float v = emb[k * NC + c]; s += v * v; }
        e2[k] = s;
    }
}

// ---------------------------------------------------------------------------
// Main: argmin over K for 128 tokens per block (8 waves x 16-token M-tile).
// Entire hi+lo codebook (256KB) + e2 (4KB) staged in LDS once.
// ---------------------------------------------------------------------------
#define LDS_LO_OFF  131072
#define LDS_E2_OFF  262144
#define LDS_TOTAL   266240

__global__ __launch_bounds__(256) void vq_argmin(const float* __restrict__ x,
                                                 const unsigned short* __restrict__ bhi,
                                                 const unsigned short* __restrict__ blo,
                                                 const float* __restrict__ e2,
                                                 float* __restrict__ idx_out) {
    extern __shared__ char smem[];
    const int tid = threadIdx.x;

    // cooperative stage: hi||lo (contiguous in ws) then e2
    {
        const uint4* src = (const uint4*)bhi;             // bhi,blo contiguous
        uint4* dst = (uint4*)smem;
        for (int i = tid; i < (2 * 131072) / 16; i += 256) dst[i] = src[i];
        const float* se = e2;
        float* de = (float*)(smem + LDS_E2_OFF);
        for (int i = tid; i < NK; i += 256) de[i] = se[i];
    }
    __syncthreads();

    const int wave  = tid >> 5;
    const int lane  = tid & 31;
    const int r     = lane & 15;
    const int g     = lane >> 4;
    const int tileM = blockIdx.x * 8 + wave;

    // ---- load A (16 tokens x 64 ch), split hi/lo, WMMA-A lane layout ------
    // A element (M,K): lanes0-15 & 16-31 both hold M = r; per lane, chunk h:
    //   vgpr v holds K = 32h + 16g + 2v + {0,1}
    const int m0   = tileM * 16 + r;                      // token id
    const int bimg = m0 >> 12;
    const int hw   = m0 & 4095;
    const int base = (bimg << 18) + hw;                   // x flat base (f32)

    BV ahi[2], alo[2];
    #pragma unroll
    for (int h = 0; h < 2; ++h) {
        #pragma unroll
        for (int j = 0; j < 16; ++j) {
            int v = j >> 1, t = j & 1;
            int c = 32 * h + 16 * g + 2 * v + t;
            float f = x[base + (c << 12)];
            split_bf16(f, ahi[h].s[j], alo[h].s[j]);
        }
    }

    float best[8];
    int   bidx[8];
    #pragma unroll
    for (int i = 0; i < 8; ++i) { best[i] = 3.4e38f; bidx[i] = 0; }

    for (int nt = 0; nt < 64; ++nt) {
        const int tb = nt * 2048 + lane * 16;
        BV bh0, bh1, bl0, bl1;
        bh0.q[0] = *(const uint4*)(smem + tb);
        bh0.q[1] = *(const uint4*)(smem + tb + 512);
        bh1.q[0] = *(const uint4*)(smem + tb + 1024);
        bh1.q[1] = *(const uint4*)(smem + tb + 1536);
        bl0.q[0] = *(const uint4*)(smem + LDS_LO_OFF + tb);
        bl0.q[1] = *(const uint4*)(smem + LDS_LO_OFF + tb + 512);
        bl1.q[0] = *(const uint4*)(smem + LDS_LO_OFF + tb + 1024);
        bl1.q[1] = *(const uint4*)(smem + LDS_LO_OFF + tb + 1536);

        const float e2v = *(const float*)(smem + LDS_E2_OFF + ((nt * 16 + r) << 2));

        v8f acc = {0.f, 0.f, 0.f, 0.f, 0.f, 0.f, 0.f, 0.f};
        // hi*hi + hi*lo + lo*hi  (bf16x2 split ~ fp32 products)
        acc = __builtin_amdgcn_wmma_f32_16x16x32_bf16(false, ahi[0].v, false, bh0.v, (short)0, acc, false, false);
        acc = __builtin_amdgcn_wmma_f32_16x16x32_bf16(false, ahi[1].v, false, bh1.v, (short)0, acc, false, false);
        acc = __builtin_amdgcn_wmma_f32_16x16x32_bf16(false, ahi[0].v, false, bl0.v, (short)0, acc, false, false);
        acc = __builtin_amdgcn_wmma_f32_16x16x32_bf16(false, ahi[1].v, false, bl1.v, (short)0, acc, false, false);
        acc = __builtin_amdgcn_wmma_f32_16x16x32_bf16(false, alo[0].v, false, bh0.v, (short)0, acc, false, false);
        acc = __builtin_amdgcn_wmma_f32_16x16x32_bf16(false, alo[1].v, false, bh1.v, (short)0, acc, false, false);

        const int n = nt * 16 + r;                        // code id of this lane's column
        #pragma unroll
        for (int i = 0; i < 8; ++i) {
            float d = e2v - 2.0f * acc[i];
            bool better = d < best[i];
            best[i] = better ? d : best[i];
            bidx[i] = better ? n : bidx[i];
        }
    }

    // D layout: lane -> N = r, VGPR i -> M = 8g + i. Reduce over the 16 lanes
    // of each half-wave (same M rows, different column residues).
    #pragma unroll
    for (int i = 0; i < 8; ++i) {
        float v = best[i];
        int   id = bidx[i];
        #pragma unroll
        for (int off = 1; off < 16; off <<= 1) {
            float ov = __shfl_xor(v, off, 32);
            int  oid = __shfl_xor(id, off, 32);
            if (ov < v || (ov == v && oid < id)) { v = ov; id = oid; }
        }
        if (r == 0) idx_out[tileM * 16 + g * 8 + i] = (float)id;
    }
}

// ---------------------------------------------------------------------------
// Gather + quant_out + per-block loss partials (deterministic two-pass).
// ---------------------------------------------------------------------------
__global__ __launch_bounds__(256) void vq_gather(const float* __restrict__ x,
                                                 const float* __restrict__ emb,
                                                 const float* __restrict__ idxf,
                                                 float* __restrict__ out,
                                                 float* __restrict__ partials) {
    __shared__ float red[256];
    int t = blockIdx.x * 256 + threadIdx.x;               // 0 .. NELE-1
    int n = t >> 6;
    int c = t & 63;
    int id = __float2int_rz(idxf[n]);
    float q  = emb[id * NC + c];
    int bimg = n >> 12;
    int hw   = n & 4095;
    int xi   = (bimg << 18) + (c << 12) + hw;
    float xv = x[xi];
    out[xi]  = q;                                         // quant_out == quant
    float d  = xv - q;
    red[threadIdx.x] = d * d;
    __syncthreads();
    for (int s = 128; s > 0; s >>= 1) {
        if (threadIdx.x < s) red[threadIdx.x] += red[threadIdx.x + s];
        __syncthreads();
    }
    if (threadIdx.x == 0) partials[blockIdx.x] = red[0];
}

__global__ __launch_bounds__(256) void vq_loss_final(const float* __restrict__ partials,
                                                     float* __restrict__ out) {
    __shared__ float red[256];
    float s = 0.f;
    for (int i = threadIdx.x; i < 16384; i += 256) s += partials[i];
    red[threadIdx.x] = s;
    __syncthreads();
    for (int k = 128; k > 0; k >>= 1) {
        if (threadIdx.x < k) red[threadIdx.x] += red[threadIdx.x + k];
        __syncthreads();
    }
    if (threadIdx.x == 0) {
        float loss = red[0] * (1.0f / (float)NELE);
        out[NELE]     = loss;                             // loss_embed
        out[NELE + 1] = loss;                             // loss_commitment (same value)
    }
}

// ---------------------------------------------------------------------------
extern "C" void kernel_launch(void* const* d_in, const int* in_sizes, int n_in,
                              void* d_out, int out_size, void* d_ws, size_t ws_size,
                              hipStream_t stream) {
    const float* x   = (const float*)d_in[0];
    const float* emb = (const float*)d_in[1];
    float* out = (float*)d_out;

    char* ws = (char*)d_ws;
    unsigned short* bhi      = (unsigned short*)(ws);               // 128 KB
    unsigned short* blo      = (unsigned short*)(ws + 131072);      // 128 KB
    float*          e2       = (float*)(ws + 262144);               //   4 KB
    float*          partials = (float*)(ws + 266240);               //  64 KB

    float* idxf = out + NELE + 2;   // [NTOK] indices stored as floats (exact)

    vq_prep_split<<<256, 256, 0, stream>>>(emb, bhi, blo);
    vq_prep_e2  <<<4,   256, 0, stream>>>(emb, e2);
    vq_argmin   <<<512, 256, LDS_TOTAL, stream>>>(x, bhi, blo, e2, idxf);
    vq_gather   <<<16384, 256, 0, stream>>>(x, emb, idxf, out, partials);
    vq_loss_final<<<1, 256, 0, stream>>>(partials, out);
}
